// Reason_58591943852624
// MI455X (gfx1250) — matrix operand order repaired and checked by reading, not compiled, not run on hardware
//
#include <hip/hip_runtime.h>
#include <hip/hip_bf16.h>
#include <stdint.h>

// ---------------- problem constants ----------------
constexpr int B_    = 2048;
constexpr int NRELS = 512;
constexpr int FEAT_ = 300;
constexpr int LAT_  = 512;
constexpr int TOPK_ = 3;
constexpr int DISK_ = 20;
constexpr float T_  = 0.07f;

// padded K dims (multiples of 32) and padded N (multiple of 16)
constexpr int KX = 608;   // concat(x,pos1): 600 -> 608
constexpr int KH = 320;   // 300 -> 320 (h1/h3/pos1/neg rows)
constexpr int KZ = 832;   // concat(z,pos1): 812 -> 832
constexpr int K5 = 512;
constexpr int NP300 = 304;

typedef __attribute__((ext_vector_type(16))) _Float16 v16h;
typedef __attribute__((ext_vector_type(8)))  _Float16 h8;
typedef __attribute__((ext_vector_type(8)))  float    v8f;

// ---------------- workspace layout (bytes, 256-aligned) ----------------
constexpr size_t AL(size_t x) { return (x + 255) & ~(size_t)255; }
constexpr size_t O_FC1T   = 0;
constexpr size_t O_FC2MUT = O_FC1T   + AL((size_t)NP300 * KX * 2);
constexpr size_t O_FC2LST = O_FC2MUT + AL((size_t)512   * KH * 2);
constexpr size_t O_FC3T   = O_FC2LST + AL((size_t)512   * KH * 2);
constexpr size_t O_FC4T   = O_FC3T   + AL((size_t)NP300 * KZ * 2);
constexpr size_t O_QI1T   = O_FC4T   + AL((size_t)NP300 * KH * 2);
constexpr size_t O_QI2T   = O_QI1T   + AL((size_t)512   * KH * 2);
constexpr size_t O_QZ1T   = O_QI2T   + AL((size_t)512   * K5 * 2);
constexpr size_t O_QZ2T   = O_QZ1T   + AL((size_t)512   * K5 * 2);
constexpr size_t O_XCAT   = O_QZ2T   + AL((size_t)512   * K5 * 2);
constexpr size_t O_POS1H  = O_XCAT   + AL((size_t)B_ * KX * 2);
constexpr size_t O_H1H    = O_POS1H  + AL((size_t)B_ * KH * 2);
constexpr size_t O_ZCAT   = O_H1H    + AL((size_t)B_ * KH * 2);
constexpr size_t O_H3H    = O_ZCAT   + AL((size_t)B_ * KZ * 2);
constexpr size_t O_ZQ     = O_H3H    + AL((size_t)B_ * KH * 2);
constexpr size_t O_HZ     = O_ZQ     + AL((size_t)B_ * K5 * 2);
constexpr size_t O_HP     = O_HZ     + AL((size_t)B_ * K5 * 2);
constexpr size_t O_NEGH   = O_HP     + AL((size_t)B_ * K5 * 2);
constexpr size_t O_HNEG   = O_NEGH   + AL((size_t)B_ * DISK_ * KH * 2);
constexpr size_t O_NEGPH  = O_HNEG   + AL((size_t)B_ * DISK_ * K5 * 2);
constexpr size_t O_XF32   = O_NEGPH  + AL((size_t)B_ * DISK_ * K5 * 2);
constexpr size_t O_MU     = O_XF32   + AL((size_t)B_ * NP300 * 4);
constexpr size_t O_LS     = O_MU     + AL((size_t)B_ * LAT_ * 4);
constexpr size_t O_RECON  = O_LS     + AL((size_t)B_ * LAT_ * 4);
constexpr size_t O_POSVEC = O_RECON  + AL((size_t)B_ * NP300 * 4);
constexpr size_t O_TOPI   = O_POSVEC + AL((size_t)B_ * LAT_ * 4);
constexpr size_t O_DISI   = O_TOPI   + AL((size_t)B_ * TOPK_ * 4);
constexpr size_t WS_USED  = O_DISI   + AL((size_t)B_ * DISK_ * 4);

// ---------------- utility kernels ----------------
__global__ void k_zero_ws(uint32_t* p, size_t n) {
    for (size_t i = blockIdx.x * (size_t)blockDim.x + threadIdx.x; i < n;
         i += (size_t)gridDim.x * blockDim.x) p[i] = 0u;
}

__global__ void k_init_out(float* out) {
    int i = blockIdx.x * blockDim.x + threadIdx.x;
    if (i < 2 + LAT_) out[i] = 0.0f;   // losses + z_rel row 0
}

// Wt[n*Kp + k] = (f16)W[k*N + n]   (pads already zeroed)
__global__ void k_wtr(const float* __restrict__ W, _Float16* __restrict__ Wt,
                      int K, int N, int Kp) {
    int total = K * N;
    for (int idx = blockIdx.x * blockDim.x + threadIdx.x; idx < total;
         idx += gridDim.x * blockDim.x) {
        int n = idx / K, k = idx - n * K;
        Wt[(size_t)n * Kp + k] = (_Float16)W[(size_t)k * N + n];
    }
}

// ---------------- top-3 / bottom-20 selection (register-only passes) ----------------
__global__ void k_topk(const float* __restrict__ logits, int* __restrict__ topi,
                       int* __restrict__ disi) {
    int b = blockIdx.x * blockDim.x + threadIdx.x;
    if (b >= B_) return;
    const float* row = logits + (size_t)b * NRELS;
    float pv = __builtin_inff(); int pi = -1;
    for (int t = 0; t < TOPK_; ++t) {
        float best = -__builtin_inff(); int bi = 0;
        for (int i = 0; i < NRELS; ++i) {
            float v = row[i];
            bool elig = (v < pv) || (v == pv && i > pi);
            if (elig && v > best) { best = v; bi = i; }
        }
        topi[b * TOPK_ + t] = bi; pv = best; pi = bi;
    }
    pv = -__builtin_inff(); pi = -1;
    for (int t = 0; t < DISK_; ++t) {
        float best = __builtin_inff(); int bi = 0;
        for (int i = 0; i < NRELS; ++i) {
            float v = row[i];
            bool elig = (v > pv) || (v == pv && i > pi);
            if (elig && v < best) { best = v; bi = i; }
        }
        disi[b * DISK_ + t] = bi; pv = best; pi = bi;
    }
}

// ---------------- gather: x, pos1, neg rows ----------------
__global__ void k_gather(const int* __restrict__ rels, const float* __restrict__ dict,
                         const int* __restrict__ topi, const int* __restrict__ disi,
                         _Float16* __restrict__ xcat, float* __restrict__ xf32,
                         _Float16* __restrict__ pos1h, _Float16* __restrict__ zcat,
                         _Float16* __restrict__ negh) {
    int b = blockIdx.x;
    const float* r0 = dict + (size_t)rels[topi[b * 3 + 0] * 2 + 1] * FEAT_;
    const float* r1 = dict + (size_t)rels[topi[b * 3 + 1] * 2 + 1] * FEAT_;
    const float* r2 = dict + (size_t)rels[topi[b * 3 + 2] * 2 + 1] * FEAT_;
    for (int f = threadIdx.x; f < FEAT_; f += blockDim.x) {
        float p0 = r0[f], p1 = r1[f], p2 = r2[f];
        float x = p0 * p1 * p2;
        xcat[(size_t)b * KX + f]          = (_Float16)x;
        xcat[(size_t)b * KX + FEAT_ + f]  = (_Float16)p0;
        xf32[(size_t)b * NP300 + f]       = x;
        pos1h[(size_t)b * KH + f]         = (_Float16)p0;
        zcat[(size_t)b * KZ + LAT_ + f]   = (_Float16)p0;
    }
    for (int t = threadIdx.x; t < DISK_ * FEAT_; t += blockDim.x) {
        int n = t / FEAT_, f = t - n * FEAT_;
        int e = rels[disi[b * DISK_ + n] * 2 + 1];
        negh[((size_t)b * DISK_ + n) * KH + f] = (_Float16)dict[(size_t)e * FEAT_ + f];
    }
}

// ---------------- 2x2-blocked WMMA GEMM: out = act(A @ Wt^T + bias) ----------------
// A:  M x Kp f16 row-major (Kp % 32 == 0, zero padded). M % 32 == 0.
// Wt: Npad x Kp f16 row-major (transposed weights, zero padded).
// Each wave computes a 32x32 output block (4 accumulators): each A fragment is
// reused across 2 B fragments and vice versa -> 2 b128 loads per WMMA.
// The n1-edge predicate is hoisted OUT of the k-loop (templated) so the hot
// loop is straight-line: no EXEC manipulation between loads and WMMAs.
// A/B: k = 16*half + [0..15] in both (contiguous); C/D layout per CDNA5 ISA 7.12.2 (wave32); EXEC all-ones at every WMMA.
__device__ __forceinline__ float apply_act(float v, int act) {
    if (act == 1)      return v > 0.0f ? v : 0.0f;
    else if (act == 2) return v > 0.0f ? v : 0.2f * v;
    else if (act == 3) return 1.0f / (1.0f + expf(-v));
    return v;
}

__device__ __forceinline__ v16h ld_frag_a(const _Float16* p) {
    // K = halfbase*2 + [0..15] contiguous
    h8 x0 = *(const h8*)(p);
    h8 x1 = *(const h8*)(p + 8);
    v16h a;
#pragma unroll
    for (int i = 0; i < 8; ++i) { a[i] = x0[i]; a[i + 8] = x1[i]; }
    return a;
}

__device__ __forceinline__ v16h ld_frag_b(const _Float16* p) {
    // K = halfbase*2 + [0..15] contiguous
    h8 x0 = *(const h8*)(p);
    h8 x1 = *(const h8*)(p + 8);
    v16h b;
#pragma unroll
    for (int i = 0; i < 8; ++i) { b[i] = x0[i]; b[i + 8] = x1[i]; }
    return b;
}

template <bool HAS_N1>
__device__ __forceinline__ void gemm_kloop(const _Float16* __restrict__ arow0,
                                           const _Float16* __restrict__ arow1,
                                           const _Float16* __restrict__ brow0,
                                           const _Float16* __restrict__ brow1,
                                           int Kp, v8f& acc00, v8f& acc01,
                                           v8f& acc10, v8f& acc11) {
    for (int k0 = 0; k0 < Kp; k0 += 32) {
        v16h a0 = ld_frag_a(arow0 + k0);
        v16h a1 = ld_frag_a(arow1 + k0);
        v16h b0 = ld_frag_b(brow0 + k0);
        acc00 = __builtin_amdgcn_wmma_f32_16x16x32_f16(false, a0, false, b0,
                                                       (short)0, acc00, false, false);
        acc10 = __builtin_amdgcn_wmma_f32_16x16x32_f16(false, a1, false, b0,
                                                       (short)0, acc10, false, false);
        if (HAS_N1) {
            v16h b1 = ld_frag_b(brow1 + k0);
            acc01 = __builtin_amdgcn_wmma_f32_16x16x32_f16(false, a0, false, b1,
                                                           (short)0, acc01, false, false);
            acc11 = __builtin_amdgcn_wmma_f32_16x16x32_f16(false, a1, false, b1,
                                                           (short)0, acc11, false, false);
        }
    }
}

__global__ void k_gemm_wmma(const _Float16* __restrict__ A, const _Float16* __restrict__ Wt,
                            const float* __restrict__ bias, float* __restrict__ outF32,
                            _Float16* __restrict__ outF16, int M, int N, int Kp,
                            int ldo32, int ldo16, int act) {
    const int lane = threadIdx.x & 31;
    const int wave = threadIdx.x >> 5;
    const int pair = blockIdx.x * 4 + wave;           // wave-uniform group of 2 n-tiles
    const int Npad = (N + 15) & ~15;
    const int n0 = pair * 32;
    if (n0 >= Npad) return;                           // whole wave exits together
    const bool has_n1 = (n0 + 16) < Npad;             // wave-uniform
    const int m0 = (int)blockIdx.y * 32;
    const int half = lane >> 4, l = lane & 15;

    const _Float16* arow0 = A  + (size_t)(m0 + l) * Kp + half * 16;
    const _Float16* arow1 = arow0 + (size_t)16 * Kp;
    const _Float16* brow0 = Wt + (size_t)(n0 + l) * Kp + half * 16;
    const _Float16* brow1 = brow0 + (size_t)16 * Kp;

    v8f acc00 = {}, acc01 = {}, acc10 = {}, acc11 = {};
    if (has_n1) gemm_kloop<true >(arow0, arow1, brow0, brow1, Kp, acc00, acc01, acc10, acc11);
    else        gemm_kloop<false>(arow0, arow1, brow0, brow1, Kp, acc00, acc01, acc10, acc11);

    // epilogue: C/D layout -> vgpr r holds M = r (+8 for upper half), N = lane%16
    const int nA = n0 + l;
    const int nB = n0 + 16 + l;
    const float bvA = (nA < N && bias) ? bias[nA] : 0.0f;
    const float bvB = (has_n1 && nB < N && bias) ? bias[nB] : 0.0f;
#pragma unroll
    for (int r = 0; r < 8; ++r) {
        int mA = m0 + half * 8 + r;
        int mB = mA + 16;
        if (nA < N) {
            float v0 = apply_act(acc00[r] + bvA, act);
            float v1 = apply_act(acc10[r] + bvA, act);
            if (outF32) { outF32[(size_t)mA * ldo32 + nA] = v0;
                          outF32[(size_t)mB * ldo32 + nA] = v1; }
            if (outF16) { outF16[(size_t)mA * ldo16 + nA] = (_Float16)v0;
                          outF16[(size_t)mB * ldo16 + nA] = (_Float16)v1; }
        }
        if (has_n1 && nB < N) {
            float v0 = apply_act(acc01[r] + bvB, act);
            float v1 = apply_act(acc11[r] + bvB, act);
            if (outF32) { outF32[(size_t)mA * ldo32 + nB] = v0;
                          outF32[(size_t)mB * ldo32 + nB] = v1; }
            if (outF16) { outF16[(size_t)mA * ldo16 + nB] = (_Float16)v0;
                          outF16[(size_t)mB * ldo16 + nB] = (_Float16)v1; }
        }
    }
}

// ---------------- VAE reparam + KL reduction ----------------
__global__ void k_vae_ew(const float* __restrict__ mu, const float* __restrict__ ls,
                         const float* __restrict__ e1, const float* __restrict__ e2,
                         _Float16* __restrict__ zcat, _Float16* __restrict__ zq,
                         float* __restrict__ out0) {
    __shared__ float sm[256];
    float acc = 0.0f;
    const size_t total = (size_t)B_ * LAT_;
    for (size_t i = blockIdx.x * (size_t)blockDim.x + threadIdx.x; i < total;
         i += (size_t)gridDim.x * blockDim.x) {
        float m = mu[i], l = ls[i];
        float s = expf(l);
        size_t b = i >> 9, d = i & 511;
        zcat[b * KZ + d] = (_Float16)(m + e1[i] * s);
        zq[i]            = (_Float16)(m + e2[i] * s);
        acc += -0.5f * (1.0f + 2.0f * l - m * m - expf(2.0f * l));
    }
    sm[threadIdx.x] = acc; __syncthreads();
    for (int s2 = 128; s2; s2 >>= 1) {
        if ((int)threadIdx.x < s2) sm[threadIdx.x] += sm[threadIdx.x + s2];
        __syncthreads();
    }
    if (threadIdx.x == 0) atomicAdd(out0, sm[0]);
}

// ---------------- recon loss reduction ----------------
__global__ void k_recon_loss(const float* __restrict__ recon, const float* __restrict__ xf32,
                             float* __restrict__ out0) {
    __shared__ float sm[256];
    float acc = 0.0f;
    const size_t total = (size_t)B_ * NP300;   // pad cols are 0-0
    for (size_t i = blockIdx.x * (size_t)blockDim.x + threadIdx.x; i < total;
         i += (size_t)gridDim.x * blockDim.x) {
        float d = recon[i] - xf32[i];
        acc += d * d;
    }
    sm[threadIdx.x] = acc; __syncthreads();
    for (int s2 = 128; s2; s2 >>= 1) {
        if ((int)threadIdx.x < s2) sm[threadIdx.x] += sm[threadIdx.x + s2];
        __syncthreads();
    }
    if (threadIdx.x == 0) atomicAdd(out0, sm[0]);
}

// ---------------- contrastive loss: one wave32 per batch row ----------------
__global__ void k_csl(const float* __restrict__ z_p, const float* __restrict__ pos_vec,
                      const _Float16* __restrict__ negph, float* __restrict__ out1) {
    int b = blockIdx.x, lane = threadIdx.x;
    const float* zr = z_p    + (size_t)b * LAT_;
    const float* pr = pos_vec + (size_t)b * LAT_;
    const _Float16* nr = negph + (size_t)b * DISK_ * LAT_;
    float dp = 0.0f;
    float dn[DISK_];
#pragma unroll
    for (int n = 0; n < DISK_; ++n) dn[n] = 0.0f;
    for (int d = lane; d < LAT_; d += 32) {
        float zv = zr[d];
        dp += zv * pr[d];
#pragma unroll
        for (int n = 0; n < DISK_; ++n) dn[n] += zv * (float)nr[(size_t)n * LAT_ + d];
    }
#pragma unroll
    for (int off = 16; off; off >>= 1) {
        dp += __shfl_xor(dp, off, 32);
#pragma unroll
        for (int n = 0; n < DISK_; ++n) dn[n] += __shfl_xor(dn[n], off, 32);
    }
    if (lane == 0) {
        float lg0 = dp / T_;
        float mx = lg0;
#pragma unroll
        for (int n = 0; n < DISK_; ++n) { float v = dn[n] / T_; mx = v > mx ? v : mx; }
        float se = expf(lg0 - mx);
#pragma unroll
        for (int n = 0; n < DISK_; ++n) se += expf(dn[n] / T_ - mx);
        atomicAdd(out1, (mx + logf(se)) - lg0);
    }
}

// ---------------- host side ----------------
static void launch_gemm(const _Float16* A, const _Float16* Wt, const float* bias,
                        float* o32, _Float16* o16, int M, int N, int Kp,
                        int ld32, int ld16, int act, hipStream_t s) {
    int ntiles = (N + 15) / 16;            // 16-col tiles
    int pairs  = (ntiles + 1) / 2;         // 32-col wave blocks
    dim3 grid((pairs + 3) / 4, M / 32);    // 4 waves per block
    k_gemm_wmma<<<grid, dim3(128), 0, s>>>(A, Wt, bias, o32, o16, M, N, Kp, ld32, ld16, act);
}

extern "C" void kernel_launch(void* const* d_in, const int* in_sizes, int n_in,
                              void* d_out, int out_size, void* d_ws, size_t ws_size,
                              hipStream_t stream) {
    const float* logits  = (const float*)d_in[0];
    const int*   rels    = (const int*)  d_in[1];
    const float* dict    = (const float*)d_in[2];
    const float* fc1_w   = (const float*)d_in[3];
    const float* fc1_b   = (const float*)d_in[4];
    const float* fc2mu_w = (const float*)d_in[5];
    const float* fc2mu_b = (const float*)d_in[6];
    const float* fc2ls_w = (const float*)d_in[7];
    const float* fc2ls_b = (const float*)d_in[8];
    const float* fc3_w   = (const float*)d_in[9];
    const float* fc3_b   = (const float*)d_in[10];
    const float* fc4_w   = (const float*)d_in[11];
    const float* fc4_b   = (const float*)d_in[12];
    const float* qi1_w   = (const float*)d_in[13];
    const float* qi1_b   = (const float*)d_in[14];
    const float* qi2_w   = (const float*)d_in[15];
    const float* qi2_b   = (const float*)d_in[16];
    const float* qz1_w   = (const float*)d_in[17];
    const float* qz1_b   = (const float*)d_in[18];
    const float* qz2_w   = (const float*)d_in[19];
    const float* qz2_b   = (const float*)d_in[20];
    const float* eps1    = (const float*)d_in[21];
    const float* eps2    = (const float*)d_in[22];
    float* out = (float*)d_out;

    char* ws = (char*)d_ws;
    _Float16* fc1t   = (_Float16*)(ws + O_FC1T);
    _Float16* fc2mut = (_Float16*)(ws + O_FC2MUT);
    _Float16* fc2lst = (_Float16*)(ws + O_FC2LST);
    _Float16* fc3t   = (_Float16*)(ws + O_FC3T);
    _Float16* fc4t   = (_Float16*)(ws + O_FC4T);
    _Float16* qi1t   = (_Float16*)(ws + O_QI1T);
    _Float16* qi2t   = (_Float16*)(ws + O_QI2T);
    _Float16* qz1t   = (_Float16*)(ws + O_QZ1T);
    _Float16* qz2t   = (_Float16*)(ws + O_QZ2T);
    _Float16* xcat   = (_Float16*)(ws + O_XCAT);
    _Float16* pos1h  = (_Float16*)(ws + O_POS1H);
    _Float16* h1h    = (_Float16*)(ws + O_H1H);
    _Float16* zcat   = (_Float16*)(ws + O_ZCAT);
    _Float16* h3h    = (_Float16*)(ws + O_H3H);
    _Float16* zq     = (_Float16*)(ws + O_ZQ);
    _Float16* hz     = (_Float16*)(ws + O_HZ);
    _Float16* hp     = (_Float16*)(ws + O_HP);
    _Float16* negh   = (_Float16*)(ws + O_NEGH);
    _Float16* hneg   = (_Float16*)(ws + O_HNEG);
    _Float16* negph  = (_Float16*)(ws + O_NEGPH);
    float* xf32   = (float*)(ws + O_XF32);
    float* mu     = (float*)(ws + O_MU);
    float* ls     = (float*)(ws + O_LS);
    float* recon  = (float*)(ws + O_RECON);
    float* posvec = (float*)(ws + O_POSVEC);
    int* topi = (int*)(ws + O_TOPI);
    int* disi = (int*)(ws + O_DISI);
    float* z_p = out + 2 + LAT_;   // z_rel rows 1..B live directly in d_out

    // 0) deterministic init
    k_zero_ws<<<4096, 256, 0, stream>>>((uint32_t*)d_ws, WS_USED / 4);
    k_init_out<<<3, 256, 0, stream>>>(out);

    // 1) weight transpose + f16 quantize (pads stay zero)
    auto tr = [&](const float* W, _Float16* Wt, int K, int N, int Kp) {
        int total = K * N;
        int g = (total + 255) / 256; if (g > 2048) g = 2048;
        k_wtr<<<g, 256, 0, stream>>>(W, Wt, K, N, Kp);
    };
    tr(fc1_w,   fc1t,   600, 300, KX);
    tr(fc2mu_w, fc2mut, 300, 512, KH);
    tr(fc2ls_w, fc2lst, 300, 512, KH);
    tr(fc3_w,   fc3t,   812, 300, KZ);
    tr(fc4_w,   fc4t,   300, 300, KH);
    tr(qi1_w,   qi1t,   300, 512, KH);
    tr(qi2_w,   qi2t,   512, 512, K5);
    tr(qz1_w,   qz1t,   512, 512, K5);
    tr(qz2_w,   qz2t,   512, 512, K5);

    // 2) selection + gathers
    k_topk<<<(B_ + 127) / 128, 128, 0, stream>>>(logits, topi, disi);
    k_gather<<<B_, 256, 0, stream>>>(rels, dict, topi, disi, xcat, xf32, pos1h, zcat, negh);

    // 3) VAE branch
    launch_gemm(xcat, fc1t,   fc1_b,   nullptr, h1h, B_, 300, KX, 0,   KH, 1, stream);
    launch_gemm(h1h,  fc2mut, fc2mu_b, mu,  nullptr, B_, 512, KH, LAT_, 0, 0, stream);
    launch_gemm(h1h,  fc2lst, fc2ls_b, ls,  nullptr, B_, 512, KH, LAT_, 0, 0, stream);
    k_vae_ew<<<2048, 256, 0, stream>>>(mu, ls, eps1, eps2, zcat, zq, out + 0);
    launch_gemm(zcat, fc3t,   fc3_b,   nullptr, h3h, B_, 300, KZ, 0,   KH, 1, stream);
    launch_gemm(h3h,  fc4t,   fc4_b,   recon, nullptr, B_, 300, KH, NP300, 0, 3, stream);
    k_recon_loss<<<1024, 256, 0, stream>>>(recon, xf32, out + 0);

    // 4) contrastive branch
    launch_gemm(zq,    qz1t, qz1_b, nullptr, hz,   B_, 512, K5, 0, K5, 2, stream);
    launch_gemm(hz,    qz2t, qz2_b, z_p, nullptr,  B_, 512, K5, K5, 0, 2, stream);
    launch_gemm(pos1h, qi1t, qi1_b, nullptr, hp,   B_, 512, KH, 0, K5, 2, stream);
    launch_gemm(hp,    qi2t, qi2_b, posvec, nullptr, B_, 512, K5, K5, 0, 2, stream);
    launch_gemm(negh,  qi1t, qi1_b, nullptr, hneg, B_ * DISK_, 512, KH, 0, K5, 2, stream);
    launch_gemm(hneg,  qi2t, qi2_b, nullptr, negph, B_ * DISK_, 512, K5, 0, K5, 2, stream);
    k_csl<<<B_, 32, 0, stream>>>(z_p, posvec, negph, out + 1);
}
